// Sampler_31061203484873
// MI455X (gfx1250) — matrix operand order repaired and checked by reading, not compiled, not run on hardware
//
#include <hip/hip_runtime.h>
#include <hip/hip_bf16.h>
#include <math.h>

// Problem dims (fixed by the reference)
#define B_DIM 128
#define D_DIM 4096
#define V_DIM 32000

typedef float  v2f  __attribute__((ext_vector_type(2)));
typedef float  v4f  __attribute__((ext_vector_type(4)));
typedef float  v8f  __attribute__((ext_vector_type(8)));
typedef __bf16 v8bf  __attribute__((ext_vector_type(8)));
typedef __bf16 v16bf __attribute__((ext_vector_type(16)));

// x ~= (float)hi + (float)lo with hi,lo bf16  ->  ~2^-16 relative operand error
__device__ __forceinline__ void split_bf16(float x, __bf16& hi, __bf16& lo)
{
    hi = (__bf16)x;
    lo = (__bf16)(x - (float)hi);
}

// ---------------------------------------------------------------------------
// Pre-pass: split H[128,4096] fp32 into bf16 hi/lo planes (2 x 1 MB, L2-
// resident). Lets the GEMM load A operands as ready-to-use bf16 b128s with
// zero per-iteration conversion cost. W (524 MB) is NOT pre-split: that
// would add HBM traffic; it is split in-registers on the fly.
// ---------------------------------------------------------------------------
__global__ __launch_bounds__(256)
void split_hidden(const float* __restrict__ H,
                  __bf16* __restrict__ Hhi,
                  __bf16* __restrict__ Hlo)
{
    const int i = blockIdx.x * 256 + threadIdx.x;
    if (i < B_DIM * D_DIM) {
        __bf16 h, l;
        split_bf16(H[i], h, l);
        Hhi[i] = h;
        Hlo[i] = l;
    }
}

// ---------------------------------------------------------------------------
// Kernel 1: logits[B,V] = (H @ W^T) / temperature, via bf16x2 split GEMM on
// V_WMMA_F32_16X16X32_BF16 (fp32 accumulate):
//     acc += Ahi*Bhi + Ahi*Blo + Alo*Bhi        (lo*lo term negligible)
//
// Roofline: W stream = 524 MB -> 22.5 us at 23.3 TB/s; AI = 64 FLOP/byte so
// the matrix pipe must sustain ~1.5 PFLOP/s. The fp32 16x16x4 pipe (~0.3-0.6
// PF) would be the bottleneck; 3x bf16 WMMAs (~101 GFLOP at ~5 PF bf16 dense
// ~ 20 us) keeps the kernel at the HBM floor with near-fp32 logits.
//
// Operand layouts (ISA 7.12.2, 16-bit, wave32):
//   A 16x32: lane<16 row=lane, VGPR0-3 = K0..7, VGPR4-7 = K16..23;
//            lane>=16 K8..15 / K24..31  -> two 16B chunks at k+half*8, +16.
//   B 32x16: lanes 0-15 col=lane hold K0..7/K16..23; lanes 16-31 K8..15/K24..31
//            -> two 8-fp32 runs at k+half*8, +16 (4 x b128), split in-reg.
// Each wave owns one 16-wide N column and ALL 8 M-tiles, so W streams from
// HBM exactly once. 500 blocks x 4 waves covers N=32000; EXEC all-1s.
// ---------------------------------------------------------------------------
__global__ __launch_bounds__(128)
void lm_head_gemm_wmma(const __bf16* __restrict__ Hhi,
                       const __bf16* __restrict__ Hlo,
                       const float*  __restrict__ W,
                       const float*  __restrict__ temperature,
                       float* __restrict__ logits)
{
    const int wave = threadIdx.x >> 5;
    const int lane = threadIdx.x & 31;
    const int half = lane >> 4;   // K half selector
    const int lm   = lane & 15;   // A row / B column within the 16-tile
    const int n0   = blockIdx.x * 64 + wave * 16;

    const float*  wb  = W   + (size_t)(n0 + lm) * D_DIM + half * 8;
    const __bf16* ahb = Hhi + (size_t)lm * D_DIM + half * 8;
    const __bf16* alb = Hlo + (size_t)lm * D_DIM + half * 8;

    v8f acc[8] = {};

    for (int k = 0; k < D_DIM; k += 32) {
        // stream-ahead hint for the 524 MB weight stream
        __builtin_prefetch(wb + k + 512, 0, 0);

        // ---- B operand: 2 x 8 consecutive fp32 (k+half*8, +16), split to bf16 hi/lo ----
        v4f w0 = *(const v4f*)(wb + k);
        v4f w1 = *(const v4f*)(wb + k + 4);
        v4f w2 = *(const v4f*)(wb + k + 16);
        v4f w3 = *(const v4f*)(wb + k + 20);
        float wf[16];
#pragma unroll
        for (int i = 0; i < 4; ++i) {
            wf[i] = w0[i]; wf[4 + i] = w1[i]; wf[8 + i] = w2[i]; wf[12 + i] = w3[i];
        }
        v16bf bhi, blo;
#pragma unroll
        for (int i = 0; i < 16; ++i) {
            __bf16 h, l;
            split_bf16(wf[i], h, l);
            bhi[i] = h;
            blo[i] = l;
        }

        // ---- A operands (pre-split bf16) + 3 WMMAs per M-tile ----
#pragma unroll
        for (int mt = 0; mt < 8; ++mt) {
            const size_t ro = (size_t)mt * 16 * D_DIM + k;
            v8bf ah0 = *(const v8bf*)(ahb + ro);
            v8bf ah1 = *(const v8bf*)(ahb + ro + 16);
            v8bf al0 = *(const v8bf*)(alb + ro);
            v8bf al1 = *(const v8bf*)(alb + ro + 16);
            v16bf ah = __builtin_shufflevector(ah0, ah1,
                0, 1, 2, 3, 4, 5, 6, 7, 8, 9, 10, 11, 12, 13, 14, 15);
            v16bf al = __builtin_shufflevector(al0, al1,
                0, 1, 2, 3, 4, 5, 6, 7, 8, 9, 10, 11, 12, 13, 14, 15);

            acc[mt] = __builtin_amdgcn_wmma_f32_16x16x32_bf16(
                false, ah, false, bhi, (short)0, acc[mt], false, false);
            acc[mt] = __builtin_amdgcn_wmma_f32_16x16x32_bf16(
                false, ah, false, blo, (short)0, acc[mt], false, false);
            acc[mt] = __builtin_amdgcn_wmma_f32_16x16x32_bf16(
                false, al, false, bhi, (short)0, acc[mt], false, false);
        }
    }

    // C/D layout: VGPR r, lane -> row M = r + (lane>=16 ? 8 : 0), col N = lane&15
#pragma unroll
    for (int mt = 0; mt < 8; ++mt) {
#pragma unroll
        for (int r = 0; r < 8; ++r) {
            const int m = mt * 16 + half * 8 + r;
            const float invt = 1.0f / temperature[m];
            logits[(size_t)m * V_DIM + n0 + lm] = acc[mt][r] * invt;
        }
    }
}

// ---------------------------------------------------------------------------
// Kernel 2: per-row top-p filtering + categorical sample (one block per row).
// Binary-search the logit threshold (no 32k sort), then Gumbel-argmax over
// the kept set == categorical over the renormalized kept distribution.
// All passes hit the L2-resident 16.4 MB logits.
// ---------------------------------------------------------------------------
__device__ __forceinline__ unsigned wang_hash(unsigned s)
{
    s = (s ^ 61u) ^ (s >> 16);
    s *= 9u;
    s ^= s >> 4;
    s *= 0x27d4eb2du;
    s ^= s >> 15;
    return s;
}

__device__ __forceinline__ float block_sum(float v, float* red, int tid)
{
    red[tid] = v;
    __syncthreads();
    for (int s = 128; s > 0; s >>= 1) {
        if (tid < s) red[tid] += red[tid + s];
        __syncthreads();
    }
    float r = red[0];
    __syncthreads();
    return r;
}

__global__ __launch_bounds__(256)
void topp_sample(const float* __restrict__ logits,
                 const float* __restrict__ top_p,
                 int* __restrict__ out)
{
    __shared__ float redA[256];
    __shared__ float redB[256];
    __shared__ int   redI[256];

    const int row = blockIdx.x;
    const int tid = threadIdx.x;
    const float* L = logits + (size_t)row * V_DIM;

    // ---- pass 1: row max / min ----
    float mx = -INFINITY, mn = INFINITY;
    for (int i = tid; i < V_DIM; i += 256) {
        float v = L[i];
        mx = fmaxf(mx, v);
        mn = fminf(mn, v);
    }
    redA[tid] = mx;
    redB[tid] = mn;
    __syncthreads();
    for (int s = 128; s > 0; s >>= 1) {
        if (tid < s) {
            redA[tid] = fmaxf(redA[tid], redA[tid + s]);
            redB[tid] = fminf(redB[tid], redB[tid + s]);
        }
        __syncthreads();
    }
    mx = redA[0];
    mn = redB[0];
    __syncthreads();

    // ---- pass 2: softmax denominator Z ----
    float part = 0.0f;
    for (int i = tid; i < V_DIM; i += 256) part += __expf(L[i] - mx);
    const float Z = block_sum(part, redA, tid);

    // ---- binary search the keep-threshold in logit space ----
    const float target = top_p[row] * Z;
    float lo = mn, hi = mx, tkeep = mn;
    for (int it = 0; it < 24; ++it) {
        const float mid = 0.5f * (lo + hi);
        float p = 0.0f;
        for (int i = tid; i < V_DIM; i += 256) {
            const float l = L[i];
            if (l >= mid) p += __expf(l - mx);
        }
        const float S = block_sum(p, redA, tid);
        if (S >= target) { tkeep = mid; lo = mid; }  // kept set still big enough
        else             { hi = mid; }
    }

    // ---- Gumbel-argmax categorical sample over the kept set ----
    float bestKey = -INFINITY;
    int   bestIdx = 0;
    for (int i = tid; i < V_DIM; i += 256) {
        const float l = L[i];
        if (l >= tkeep) {
            const unsigned h = wang_hash((unsigned)(row * V_DIM + i) ^ 0x9E3779B9u);
            const float u = ((h >> 8) + 0.5f) * (1.0f / 16777216.0f);  // (0,1)
            const float g = -__logf(-__logf(u));                        // Gumbel(0,1)
            const float key = (l - mx) + g;
            if (key > bestKey) { bestKey = key; bestIdx = i; }
        }
    }
    redA[tid] = bestKey;
    redI[tid] = bestIdx;
    __syncthreads();
    for (int s = 128; s > 0; s >>= 1) {
        if (tid < s && redA[tid + s] > redA[tid]) {
            redA[tid] = redA[tid + s];
            redI[tid] = redI[tid + s];
        }
        __syncthreads();
    }
    if (tid == 0) out[row] = redI[0];
}

// ---------------------------------------------------------------------------
// Launch. Inputs (setup_inputs order): hidden_states, embd_weight,
// temperature, top_p. Workspace: logits (16.384 MB) + Hhi/Hlo bf16 planes
// (1 MB each).
// ---------------------------------------------------------------------------
extern "C" void kernel_launch(void* const* d_in, const int* in_sizes, int n_in,
                              void* d_out, int out_size, void* d_ws, size_t ws_size,
                              hipStream_t stream)
{
    (void)in_sizes; (void)n_in; (void)out_size; (void)ws_size;

    const float* H    = (const float*)d_in[0];
    const float* W    = (const float*)d_in[1];
    const float* temp = (const float*)d_in[2];
    const float* topp = (const float*)d_in[3];

    float*  logits = (float*)d_ws;
    __bf16* Hhi    = (__bf16*)((char*)d_ws + (size_t)B_DIM * V_DIM * sizeof(float));
    __bf16* Hlo    = Hhi + (size_t)B_DIM * D_DIM;
    int*    out    = (int*)d_out;

    // split H into bf16 hi/lo planes (2048 blocks x 256 threads)
    split_hidden<<<(B_DIM * D_DIM + 255) / 256, 256, 0, stream>>>(H, Hhi, Hlo);

    // 32000 / (4 waves * 16 cols) = 500 blocks, 128 threads (4 wave32)
    lm_head_gemm_wmma<<<V_DIM / 64, 128, 0, stream>>>(Hhi, Hlo, W, temp, logits);

    // one workgroup per batch row
    topp_sample<<<B_DIM, 256, 0, stream>>>(logits, topp, out);
}
